// DecoderWithAttention_38757784879662
// MI455X (gfx1250) — matrix operand order repaired and checked by reading, not compiled, not run on hardware
//
#include <hip/hip_runtime.h>
#include <hip/hip_bf16.h>
#include <math.h>

#define B_     128
#define P_     196
#define E_     512
#define H_     512
#define A_     256
#define V_     30000
#define TSTEPS 20          // T-1 decode steps
#define KCAT   (2*E_+H_)   // 1536
#define NG     (4*H_)      // 2048

typedef __bf16 bf16;
typedef __attribute__((ext_vector_type(16))) __bf16 v16bf;
typedef __attribute__((ext_vector_type(8)))  __bf16 v8bf;
typedef __attribute__((ext_vector_type(8)))  float  v8f;

// ---------------- WMMA fragment loaders (wave32, 16x16x32 bf16) ----------------

// A: row-major M x K tile at (m0,k0). ISA layout: lanes 0-15 -> M=lane, K-octets
// {kh..kh+7, 16+kh..16+kh+7} with kh = (lane>=16)*8.
__device__ __forceinline__ v16bf load_a_frag(const bf16* __restrict__ A, int lda,
                                             int m0, int k0, int lane) {
  int m  = lane & 15;
  int kh = (lane >> 4) * 8;
  const bf16* ap = A + (size_t)(m0 + m) * lda + k0;
  v8bf lo = *(const v8bf*)(ap + kh);        // 16B load
  v8bf hi = *(const v8bf*)(ap + 16 + kh);   // 16B load
  v16bf a;
#pragma unroll
  for (int j = 0; j < 8; ++j) { a[j] = lo[j]; a[8 + j] = hi[j]; }
  return a;
}

// B = W^T where W is row-major N x K: B[k][n] = W[n][k]. ISA layout: column
// n = lane&15; K-octets {kh..kh+7, 16+kh..16+kh+7} with kh = (lane>=16)*8
// (two 16B loads).
__device__ __forceinline__ v16bf load_bt_frag(const bf16* __restrict__ W, int ldw,
                                              int n0, int k0, int lane) {
  int n  = lane & 15;
  int kh = (lane >> 4) * 8;
  const bf16* bp = W + (size_t)(n0 + n) * ldw + k0 + kh;
  v8bf lo = *(const v8bf*)(bp);
  v8bf hi = *(const v8bf*)(bp + 16);
  v16bf b;
#pragma unroll
  for (int j = 0; j < 8; ++j) { b[j] = lo[j]; b[8 + j] = hi[j]; }
  return b;
}

// ------- GEMM: out(f32) = A(bf16,MxK) * W(bf16,NxK)^T + bias -------
// Each wave computes MT stacked 16x16 tiles (16*MT rows x 16 cols): the B
// fragment is loaded ONCE per k-step and reused across MT independent
// accumulators -> weight matrix streamed from L2 exactly once when 16*MT == M.
template <int MT>
__global__ void k_gemm_bt(const bf16* __restrict__ A, const bf16* __restrict__ W,
                          const float* __restrict__ bias, float* __restrict__ out,
                          int M, int N, int K, long ldo) {
  int wave = threadIdx.x >> 5;
  int lane = threadIdx.x & 31;
  int n_tile = blockIdx.x * 8 + wave;       // wave-uniform -> safe early exit
  int m0 = blockIdx.y * (16 * MT);
  int n0 = n_tile * 16;
  if (n0 >= N || m0 >= M) return;

  v8f acc[MT];
#pragma unroll
  for (int mi = 0; mi < MT; ++mi) acc[mi] = (v8f){};

  for (int k0 = 0; k0 < K; k0 += 32) {
    v16bf b = load_bt_frag(W, K, n0, k0, lane);   // one B load, MT-fold reuse
#pragma unroll
    for (int mi = 0; mi < MT; ++mi) {
      v16bf a = load_a_frag(A, K, m0 + 16 * mi, k0, lane);
      acc[mi] = __builtin_amdgcn_wmma_f32_16x16x32_bf16(false, a, false, b,
                                                        (short)0, acc[mi], false, false);
    }
  }
  // C/D layout: col = n0 + (lane&15); lanes 0-15 -> rows +0..7, lanes 16-31 -> +8..15
  int n  = lane & 15;
  int mo = (lane >> 4) * 8;
  float bn = bias ? bias[n0 + n] : 0.0f;
#pragma unroll
  for (int mi = 0; mi < MT; ++mi) {
    int mr = m0 + 16 * mi + mo;
#pragma unroll
    for (int j = 0; j < 8; ++j)
      out[(size_t)(mr + j) * ldo + n0 + n] = acc[mi][j] + bn;
  }
}

// ---------------- One-time prep kernels ----------------

__global__ void k_f32_to_bf16(const float* __restrict__ s, bf16* __restrict__ d, int n) {
  int i = blockIdx.x * blockDim.x + threadIdx.x;
  if (i < n) d[i] = (bf16)s[i];
}

// s: K x N (row-major), d: N x K bf16 (so it can be used as a W^T operand)
__global__ void k_transpose_to_bf16(const float* __restrict__ s, bf16* __restrict__ d,
                                    int K, int N) {
  int i = blockIdx.x * blockDim.x + threadIdx.x;
  if (i < N * K) {
    int n = i / K, k = i % K;
    d[i] = (bf16)s[(size_t)k * N + n];
  }
}

// W_cat[n][k] = k<1024 ? W_ih[n][k] : W_hh[n][k-1024]  (n<2048, k<1536)
__global__ void k_build_wcat(const float* __restrict__ Wih, const float* __restrict__ Whh,
                             bf16* __restrict__ d) {
  int i = blockIdx.x * blockDim.x + threadIdx.x;
  if (i < NG * KCAT) {
    int n = i / KCAT, k = i % KCAT;
    float v = (k < 2 * E_) ? Wih[(size_t)n * (2 * E_) + k]
                           : Whh[(size_t)n * H_ + (k - 2 * E_)];
    d[i] = (bf16)v;
  }
}

__global__ void k_bias_cat(const float* __restrict__ bih, const float* __restrict__ bhh,
                           float* __restrict__ d) {
  int i = blockIdx.x * blockDim.x + threadIdx.x;
  if (i < NG) d[i] = bih[i] + bhh[i];
}

// x_emb_bf16[b][t][e] = emb[captions[b][t]][e], t = 0..TSTEPS-1
__global__ void k_embed(const float* __restrict__ emb, const int* __restrict__ cap,
                        bf16* __restrict__ xe) {
  int i = blockIdx.x * blockDim.x + threadIdx.x;
  if (i < B_ * TSTEPS * E_) {
    int e = i % E_;
    int bt = i / E_;
    int t = bt % TSTEPS, b = bt / TSTEPS;
    int tok = cap[b * (TSTEPS + 1) + t];
    xe[i] = (bf16)emb[(size_t)tok * E_ + e];
  }
}

__global__ void k_init_state(float* __restrict__ c, bf16* __restrict__ hb) {
  int i = blockIdx.x * blockDim.x + threadIdx.x;
  if (i < B_ * H_) { c[i] = 0.0f; hb[i] = (bf16)0.0f; }
}

// ---------------- Fused attention (one block per batch row, 256 threads) -------
// e[p] = b_full + sum_a relu(enc_proj[b,p,a] + dec_att[b,a]) * w_full[a]
// alpha = softmax_p(e);  context[e] = sum_p alpha[p]*enc[b,p,e]
// Also packs x_cat[b] = [ x_emb(t) | context | h_prev ] in bf16.
__global__ void k_attention(const float* __restrict__ enc_proj,
                            const float* __restrict__ enc,
                            const float* __restrict__ dec_att,
                            const float* __restrict__ w_full,
                            const float* __restrict__ b_full,
                            const bf16* __restrict__ x_emb,
                            const bf16* __restrict__ h_bf,
                            bf16* __restrict__ x_cat, int t) {
  int b = blockIdx.x;
  int tid = threadIdx.x;                 // 256 threads = 8 waves
  __shared__ float sdec[A_];
  __shared__ float sw[A_];
  __shared__ float se[256];
  __shared__ float red[256];

  sdec[tid] = dec_att[(size_t)b * A_ + tid];
  sw[tid]   = w_full[tid];
  __syncthreads();

  float ev = -INFINITY;
  if (tid < P_) {
    const float* ep = enc_proj + ((size_t)b * P_ + tid) * A_;
    float acc = b_full[0];
#pragma unroll 4
    for (int a = 0; a < A_; ++a) {
      float v = ep[a] + sdec[a];
      acc += (v > 0.0f ? v : 0.0f) * sw[a];
    }
    ev = acc;
  }
  red[tid] = ev; __syncthreads();
  for (int s = 128; s > 0; s >>= 1) {
    if (tid < s) red[tid] = fmaxf(red[tid], red[tid + s]);
    __syncthreads();
  }
  float mx = red[0]; __syncthreads();

  float ex = (tid < P_) ? __expf(ev - mx) : 0.0f;
  se[tid] = ex; red[tid] = ex; __syncthreads();
  for (int s = 128; s > 0; s >>= 1) {
    if (tid < s) red[tid] += red[tid + s];
    __syncthreads();
  }
  float inv = 1.0f / red[0];
  __syncthreads();

  // context + pack x_cat = [xt | context | h_prev]
  // (enc read coalesced: lane e for fixed p -> consecutive addresses)
  for (int e = tid; e < E_; e += 256) {
    float acc = 0.0f;
    const float* base = enc + (size_t)b * P_ * E_ + e;
    for (int p = 0; p < P_; ++p) acc += se[p] * base[(size_t)p * E_];
    x_cat[(size_t)b * KCAT + E_ + e]     = (bf16)(acc * inv);
    x_cat[(size_t)b * KCAT + e]          = x_emb[((size_t)b * TSTEPS + t) * E_ + e];
    x_cat[(size_t)b * KCAT + 2 * E_ + e] = h_bf[(size_t)b * H_ + e];
  }
}

// ---------------- LSTM pointwise ----------------
__global__ void k_lstm(const float* __restrict__ gates, float* __restrict__ c,
                       bf16* __restrict__ hb) {
  int i = blockIdx.x * blockDim.x + threadIdx.x;
  if (i >= B_ * H_) return;
  int b = i / H_, j = i % H_;
  const float* g = gates + (size_t)b * NG;
  float ig = g[j], fg = g[H_ + j], gg = g[2 * H_ + j], og = g[3 * H_ + j];
  float si = 1.0f / (1.0f + __expf(-ig));
  float sf = 1.0f / (1.0f + __expf(-fg));
  float so = 1.0f / (1.0f + __expf(-og));
  float cn = sf * c[i] + si * tanhf(gg);
  float hn = so * tanhf(cn);
  c[i] = cn;
  hb[i] = (bf16)hn;
}

// ---------------- Host launcher ----------------
extern "C" void kernel_launch(void* const* d_in, const int* in_sizes, int n_in,
                              void* d_out, int out_size, void* d_ws, size_t ws_size,
                              hipStream_t stream) {
  (void)in_sizes; (void)n_in; (void)out_size; (void)ws_size;
  const float* enc   = (const float*)d_in[0];
  const int*   cap   = (const int*)d_in[1];
  const float* emb   = (const float*)d_in[2];
  const float* Wenc  = (const float*)d_in[3];
  const float* benc  = (const float*)d_in[4];
  const float* Wdec  = (const float*)d_in[5];
  const float* bdec  = (const float*)d_in[6];
  const float* wfull = (const float*)d_in[7];
  const float* bfull = (const float*)d_in[8];
  const float* Wih   = (const float*)d_in[9];
  const float* bih   = (const float*)d_in[10];
  const float* Whh   = (const float*)d_in[11];
  const float* bhh   = (const float*)d_in[12];
  const float* Wfc   = (const float*)d_in[13];
  const float* bfc   = (const float*)d_in[14];
  float* out = (float*)d_out;

  size_t off = 0;
  char* wsb = (char*)d_ws;
  auto carve = [&](size_t bytes) -> void* {
    off = (off + 255) & ~(size_t)255;
    void* p = (void*)(wsb + off);
    off += bytes;
    return p;
  };

  bf16*  enc_bf  = (bf16*) carve((size_t)B_ * P_ * E_ * 2);      // 25.7 MB
  bf16*  wenc_t  = (bf16*) carve((size_t)A_ * E_ * 2);
  bf16*  wdec_t  = (bf16*) carve((size_t)A_ * H_ * 2);
  bf16*  wcat    = (bf16*) carve((size_t)NG * KCAT * 2);         // 6.3 MB
  bf16*  wfc_b   = (bf16*) carve((size_t)V_ * H_ * 2);           // 30.7 MB (L2-resident)
  bf16*  xemb    = (bf16*) carve((size_t)B_ * TSTEPS * E_ * 2);
  bf16*  xcat    = (bf16*) carve((size_t)B_ * KCAT * 2);
  bf16*  hb      = (bf16*) carve((size_t)B_ * H_ * 2);
  float* encproj = (float*)carve((size_t)B_ * P_ * A_ * 4);      // 25.7 MB
  float* decatt  = (float*)carve((size_t)B_ * A_ * 4);
  float* gatesb  = (float*)carve((size_t)B_ * NG * 4);
  float* cstate  = (float*)carve((size_t)B_ * H_ * 4);
  float* biascat = (float*)carve((size_t)NG * 4);

  const int THR = 256;
  // ---- one-time prep ----
  k_f32_to_bf16<<<(B_ * P_ * E_ + THR - 1) / THR, THR, 0, stream>>>(enc, enc_bf, B_ * P_ * E_);
  k_f32_to_bf16<<<(V_ * H_ + THR - 1) / THR, THR, 0, stream>>>(Wfc, wfc_b, V_ * H_);
  k_transpose_to_bf16<<<(A_ * E_ + THR - 1) / THR, THR, 0, stream>>>(Wenc, wenc_t, E_, A_);
  k_transpose_to_bf16<<<(A_ * H_ + THR - 1) / THR, THR, 0, stream>>>(Wdec, wdec_t, H_, A_);
  k_build_wcat<<<(NG * KCAT + THR - 1) / THR, THR, 0, stream>>>(Wih, Whh, wcat);
  k_bias_cat<<<(NG + THR - 1) / THR, THR, 0, stream>>>(bih, bhh, biascat);
  k_embed<<<(B_ * TSTEPS * E_ + THR - 1) / THR, THR, 0, stream>>>(emb, cap, xemb);
  k_init_state<<<(B_ * H_ + THR - 1) / THR, THR, 0, stream>>>(cstate, hb);

  // ---- enc_proj = encoder_out @ W_enc_att + b : M=25088 N=256 K=512 (MT=4) ----
  k_gemm_bt<4><<<dim3((A_ / 16 + 7) / 8, (B_ * P_) / 64), THR, 0, stream>>>(
      enc_bf, wenc_t, benc, encproj, B_ * P_, A_, E_, A_);

  // ---- decode loop (M=128 GEMMs use MT=8 -> weights streamed once) ----
  for (int t = 0; t < TSTEPS; ++t) {
    // dec_att = h @ W_dec_att + b : 128 x 256 x 512
    k_gemm_bt<8><<<dim3((A_ / 16 + 7) / 8, 1), THR, 0, stream>>>(
        hb, wdec_t, bdec, decatt, B_, A_, H_, A_);
    // attention + softmax + context + pack x_cat
    k_attention<<<B_, THR, 0, stream>>>(encproj, enc, decatt, wfull, bfull,
                                        xemb, hb, xcat, t);
    // gates = x_cat @ W_cat^T + (b_ih+b_hh) : 128 x 2048 x 1536
    k_gemm_bt<8><<<dim3((NG / 16 + 7) / 8, 1), THR, 0, stream>>>(
        xcat, wcat, biascat, gatesb, B_, NG, KCAT, NG);
    // LSTM cell pointwise -> h (bf16), c (f32)
    k_lstm<<<(B_ * H_ + THR - 1) / THR, THR, 0, stream>>>(gatesb, cstate, hb);
    // logits = h_new @ W_fc^T + b_fc : 128 x 30000 x 512, write out[:, t, :]
    k_gemm_bt<8><<<dim3((V_ / 16 + 7) / 8, 1), THR, 0, stream>>>(
        hb, wfc_b, bfc, out + (size_t)t * V_, B_, V_, H_, (long)TSTEPS * V_);
  }
}
